// GraphDownsamplingBlock_21071109554685
// MI455X (gfx1250) — matrix operand order repaired and checked by reading, not compiled, not run on hardware
//
#include <hip/hip_runtime.h>
#include <stdint.h>

// ---------------- static problem geometry ----------------
#define GH 495
#define GW 436
#define NN (GH * GW)            // 215820 nodes
#define PH 248                  // ceil(495/2)
#define PW 218                  // 436/2
#define NP (PH * PW)            // 54064 pooled nodes
#define EN (494 * 436)          // 215384 north edges
#define EE (495 * 435)          // 215325 east edges
#define ES (494 * 436)          // 215384 south edges
#define EW (495 * 435)          // 215325 west edges
#define T64 ((NN + 63) / 64)    // 3373 macro-tiles of 64 nodes

// kept pooled edges per direction
#define KN (247 * 218)          // 53846
#define KE (248 * 217)          // 53816
#define KS (247 * 218)          // 53846
#define KW (248 * 217)          // 53816

// ---------------- output layout (floats) ----------------
#define O_XP   ((size_t)0)
#define O_POS  (O_XP  + (size_t)NP * 128)
#define O_EIN  (O_POS + (size_t)NP * 2)
#define O_EIE  (O_EIN + (size_t)KN * 2)
#define O_EIS  (O_EIE + (size_t)KE * 2)
#define O_EIW  (O_EIS + (size_t)KS * 2)
#define O_EFN  (O_EIW + (size_t)KW * 2)
#define O_EFE  (O_EFN + (size_t)KN * 32)
#define O_EFS  (O_EFE + (size_t)KE * 32)
#define O_EFW  (O_EFS + (size_t)KS * 32)

// ---------------- workspace layout (bytes) ----------------
#define WS_XQ  ((size_t)0)                              // N x 128 bf16
#define WS_AGG (WS_XQ  + (size_t)NN * 128 * 2)          // N x 128 bf16
#define WS_X2  (WS_AGG + (size_t)NN * 128 * 2)          // N x 128 f32
#define WS_QN  (WS_X2  + (size_t)NN * 128 * 4)          // EN x 32 f32
#define WS_QE  (WS_QN  + (size_t)EN * 32 * 4)
#define WS_QS  (WS_QE  + (size_t)EE * 32 * 4)
#define WS_QW  (WS_QS  + (size_t)ES * 32 * 4)

typedef unsigned short u16;
typedef __attribute__((ext_vector_type(8)))  float  v8f;
typedef __attribute__((ext_vector_type(16))) __bf16 v16bf;

// fp32 -> bf16 via clang's native conversion (hardware cvt on gfx1250)
__device__ __forceinline__ u16 f2bf(float f) {
  return __builtin_bit_cast(u16, (__bf16)f);
}
__device__ __forceinline__ unsigned pkbf(float a, float b) {
  return (unsigned)f2bf(a) | ((unsigned)f2bf(b) << 16);
}

// A fragment: two contiguous 8-element (16B) runs per lane.
// 16-bit A 16x32 layout: lanes0-15 hold K={0..7,16..23}, lanes16-31 +8.
__device__ __forceinline__ v16bf frag2(const u16* p0, const u16* p1) {
  v16bf r;
#pragma unroll
  for (int i = 0; i < 8; ++i) {
    r[i]     = __builtin_bit_cast(__bf16, p0[i]);
    r[8 + i] = __builtin_bit_cast(__bf16, p1[i]);
  }
  return r;
}
// B fragment prepacked contiguously in LDS (32B per lane).
__device__ __forceinline__ v16bf frag16(const u16* p) {
  v16bf r;
#pragma unroll
  for (int i = 0; i < 16; ++i) r[i] = __builtin_bit_cast(__bf16, p[i]);
  return r;
}
__device__ __forceinline__ v8f wmma_bf16(v16bf a, v16bf b, v8f c) {
  return __builtin_amdgcn_wmma_f32_16x16x32_bf16(false, a, false, b, (short)0, c,
                                                 false, false);
}

// CDNA5 async global->LDS copy (16B per lane, ASYNCcnt-tracked).
// Low 32 bits of a generic LDS pointer are the LDS byte address.
__device__ __forceinline__ void async_ld_b128(unsigned lds_addr, const void* g) {
  asm volatile("global_load_async_to_lds_b128 %0, %1, off"
               :: "v"(lds_addr), "v"((unsigned long long)(uintptr_t)g)
               : "memory");
}
__device__ __forceinline__ void wait_async0() {
  asm volatile("s_wait_asynccnt 0" ::: "memory");
}

// =====================================================================
// Kernel 1: xq = relu(x @ W_emb_all)   [N,128]@[128,128] -> bf16
// 64-row macro-tiles; wave (mt,nh) computes 16x64; 16 WMMA / wave / tile
// =====================================================================
__global__ __launch_bounds__(256) void k_embed(const float* __restrict__ x,
                                               const float* __restrict__ Wemb,
                                               u16* __restrict__ xq) {
  __shared__ u16 sW[16384];  // [nc8][kc4][lane32][e16]
  __shared__ u16 sA[8192];   // 64 x 128 bf16
  __shared__ u16 sO[8192];   // 64 x 128 bf16
  const int t = threadIdx.x, lane = t & 31, w = t >> 5;
  const int mt = w >> 1, nh = w & 1;

  for (int i = t; i < 16384; i += 256) {
    int e = i & 15, l = (i >> 4) & 31, kc = (i >> 9) & 3, nc = i >> 11;
    int k = kc * 32 + ((l >> 4) << 3) + (e & 7) + ((e >> 3) << 4);
    int j = nc * 16 + (l & 15);
    sW[i] = f2bf(Wemb[((j >> 5) * 128 + k) * 32 + (j & 31)]);
  }
  __syncthreads();

  const int m = lane & 15, off = (lane >> 4) * 8;
  for (int tile = blockIdx.x; tile < T64; tile += gridDim.x) {
    int base = tile * 64;
    int nb = base + (int)gridDim.x * 64;
    if (nb < NN) __builtin_prefetch(x + (size_t)nb * 128 + t * 32, 0, 0);

    for (int i = 0; i < 4; ++i) {  // stage A: 1024 x (8 bf16) groups
      int g = t + i * 256;
      int r = g >> 4, gc = g & 15;
      int node = base + r;
      uint4 pk = {0, 0, 0, 0};
      if (node < NN) {
        const float4* p = (const float4*)(x + (size_t)node * 128 + gc * 8);
        float4 a = p[0], b = p[1];
        pk.x = pkbf(a.x, a.y); pk.y = pkbf(a.z, a.w);
        pk.z = pkbf(b.x, b.y); pk.w = pkbf(b.z, b.w);
      }
      ((uint4*)sA)[g] = pk;
    }
    __syncthreads();

    v8f acc[4] = {{}, {}, {}, {}};
    const u16* arow = &sA[(mt * 16 + m) * 128];
#pragma unroll
    for (int kc = 0; kc < 4; ++kc) {
      v16bf a = frag2(arow + kc * 32 + off, arow + kc * 32 + off + 16);
#pragma unroll
      for (int j = 0; j < 4; ++j) {
        v16bf b = frag16(&sW[(((nh * 4 + j) * 4 + kc) * 32 + lane) * 16]);
        acc[j] = wmma_bf16(a, b, acc[j]);
      }
    }
#pragma unroll
    for (int j = 0; j < 4; ++j)
#pragma unroll
      for (int v = 0; v < 8; ++v) {
        int row = v + ((lane >> 4) << 3);
        sO[(mt * 16 + row) * 128 + (nh * 4 + j) * 16 + (lane & 15)] =
            f2bf(fmaxf(acc[j][v], 0.f));
      }
    __syncthreads();

    for (int i = 0; i < 4; ++i) {  // coalesced bf16 store
      int g = t + i * 256;
      int node = base + (g >> 4);
      if (node < NN)
        ((uint4*)xq)[(size_t)node * 16 + (g & 15)] = ((const uint4*)sO)[g];
    }
    __syncthreads();
  }
}

// =====================================================================
// Kernel 2: per-direction edge GEMM + scatter (async LDS gathers)
// =====================================================================
__global__ __launch_bounds__(256) void k_edge(const float* __restrict__ ef,
                                              const int* __restrict__ ei,
                                              const u16* __restrict__ xq,
                                              const float* __restrict__ Wq,
                                              u16* __restrict__ aggs,
                                              float* __restrict__ qedge,
                                              int E, int dir, int nTiles) {
  __shared__ u16 sW[3072];   // [nc2][kc3][lane32][e16]
  __shared__ u16 sA[6144];   // 64 x 96
  __shared__ float sC[2048]; // 64 x 32
  const int t = threadIdx.x, lane = t & 31, w = t >> 5;

  for (int i = t; i < 3072; i += 256) {
    int e = i & 15, l = (i >> 4) & 31, rest = i >> 9;
    int kc = rest % 3, nc = rest / 3;
    int k = kc * 32 + ((l >> 4) << 3) + (e & 7) + ((e >> 3) << 4);  // k < 96
    sW[i] = f2bf(Wq[k * 32 + nc * 16 + (l & 15)]);
  }
  __syncthreads();

  const int m = lane & 15, off = (lane >> 4) * 8;
  for (int tile = blockIdx.x; tile < nTiles; tile += gridDim.x) {
    int ebase = tile * 64;
    {  // edge feats -> cols 0..31 (one 16B group per thread)
      int r = t >> 2, gc = t & 3;
      int e = ebase + r;
      uint4 pk = {0, 0, 0, 0};
      if (e < E) {
        const float4* p = (const float4*)(ef + (size_t)e * 32 + gc * 8);
        float4 a = p[0], b = p[1];
        pk.x = pkbf(a.x, a.y); pk.y = pkbf(a.z, a.w);
        pk.z = pkbf(b.x, b.y); pk.w = pkbf(b.z, b.w);
      }
      ((uint4*)sA)[r * 12 + gc] = pk;
    }
    {  // xq[src]/xq[dst] -> cols 32..63 / 64..95 via async global->LDS
      int r = t >> 2, seg = t & 3;
      int e = ebase + r;
      if (e < E) {
        int s = ei[(size_t)e * 2 + 0];
        int d = ei[(size_t)e * 2 + 1];
        unsigned l0 = (unsigned)(uintptr_t)&sA[r * 96 + 32 + seg * 8];
        unsigned l1 = (unsigned)(uintptr_t)&sA[r * 96 + 64 + seg * 8];
        async_ld_b128(l0, xq + (size_t)s * 128 + dir * 32 + seg * 8);
        async_ld_b128(l1, xq + (size_t)d * 128 + dir * 32 + seg * 8);
      } else {
        uint4 z = {0, 0, 0, 0};
        ((uint4*)sA)[r * 12 + 4 + seg] = z;
        ((uint4*)sA)[r * 12 + 8 + seg] = z;
      }
    }
    wait_async0();
    __syncthreads();

    int mt = w >> 1, nt = w & 1;
    const u16* arow = &sA[(mt * 16 + m) * 96];
    v8f acc = {};
#pragma unroll
    for (int kc = 0; kc < 3; ++kc) {
      v16bf a = frag2(arow + kc * 32 + off, arow + kc * 32 + off + 16);
      v16bf b = frag16(&sW[(((nt * 3) + kc) * 32 + lane) * 16]);
      acc = wmma_bf16(a, b, acc);
    }
#pragma unroll
    for (int v = 0; v < 8; ++v) {
      int row = v + ((lane >> 4) << 3);
      sC[(mt * 16 + row) * 32 + nt * 16 + (lane & 15)] = fmaxf(acc[v], 0.f);
    }
    __syncthreads();

    {  // writeback: qedge f32 (coalesced) + unique-dst bf16 scatter
      int r = t >> 2, seg = t & 3;
      int e = ebase + r;
      if (e < E) {
        const float4* sc4 = (const float4*)sC;
        float4 f0 = sc4[r * 8 + seg * 2 + 0];
        float4 f1 = sc4[r * 8 + seg * 2 + 1];
        ((float4*)qedge)[(size_t)e * 8 + seg * 2 + 0] = f0;
        ((float4*)qedge)[(size_t)e * 8 + seg * 2 + 1] = f1;
        int d = ei[(size_t)e * 2 + 1];
        uint4 pk;
        pk.x = pkbf(f0.x * 0.25f, f0.y * 0.25f);
        pk.y = pkbf(f0.z * 0.25f, f0.w * 0.25f);
        pk.z = pkbf(f1.x * 0.25f, f1.y * 0.25f);
        pk.w = pkbf(f1.z * 0.25f, f1.w * 0.25f);
        ((uint4*)aggs)[(size_t)d * 16 + dir * 4 + seg] = pk;
      }
    }
    __syncthreads();
  }
}

// =====================================================================
// Kernel 3: x2 = relu([x | aggs] @ W_node)   [N,256]@[256,128] -> f32
// 64-row macro-tiles; 32 WMMA / wave / tile
// =====================================================================
__global__ __launch_bounds__(256) void k_node(const float* __restrict__ x,
                                              const u16* __restrict__ aggs,
                                              const float* __restrict__ Wn,
                                              float* __restrict__ x2) {
  __shared__ u16 sW[32768];  // [nc8][kc8][lane32][e16]  (64 KB)
  __shared__ u16 sA[16384];  // 64 x 256 bf16 (32 KB)
  __shared__ float sO[8192]; // 64 x 128 f32  (32 KB)
  const int t = threadIdx.x, lane = t & 31, w = t >> 5;
  const int mt = w >> 1, nh = w & 1;

  for (int i = t; i < 32768; i += 256) {
    int e = i & 15, l = (i >> 4) & 31, kc = (i >> 9) & 7, nc = (i >> 12) & 7;
    int k = kc * 32 + ((l >> 4) << 3) + (e & 7) + ((e >> 3) << 4);  // k < 256
    sW[i] = f2bf(Wn[k * 128 + nc * 16 + (l & 15)]);
  }
  __syncthreads();

  const int m = lane & 15, off = (lane >> 4) * 8;
  for (int tile = blockIdx.x; tile < T64; tile += gridDim.x) {
    int base = tile * 64;
    int nb = base + (int)gridDim.x * 64;
    if (nb < NN) {
      __builtin_prefetch(x + (size_t)nb * 128 + t * 32, 0, 0);
      __builtin_prefetch(aggs + (size_t)nb * 128 + t * 64, 0, 0);
    }
    for (int i = 0; i < 8; ++i) {  // stage A: 2048 x 16B groups
      int g = t + i * 256;
      int r = g >> 5, gc = g & 31;
      int node = base + r;
      uint4 pk = {0, 0, 0, 0};
      if (node < NN) {
        if (gc < 16) {
          const float4* p = (const float4*)(x + (size_t)node * 128 + gc * 8);
          float4 a = p[0], b = p[1];
          pk.x = pkbf(a.x, a.y); pk.y = pkbf(a.z, a.w);
          pk.z = pkbf(b.x, b.y); pk.w = pkbf(b.z, b.w);
        } else {
          pk = ((const uint4*)aggs)[(size_t)node * 16 + (gc - 16)];
        }
      }
      ((uint4*)sA)[g] = pk;
    }
    __syncthreads();

    v8f acc[4] = {{}, {}, {}, {}};
    const u16* arow = &sA[(mt * 16 + m) * 256];
#pragma unroll
    for (int kc = 0; kc < 8; ++kc) {
      v16bf a = frag2(arow + kc * 32 + off, arow + kc * 32 + off + 16);
#pragma unroll
      for (int j = 0; j < 4; ++j) {
        v16bf b = frag16(&sW[(((nh * 4 + j) * 8 + kc) * 32 + lane) * 16]);
        acc[j] = wmma_bf16(a, b, acc[j]);
      }
    }
#pragma unroll
    for (int j = 0; j < 4; ++j)
#pragma unroll
      for (int v = 0; v < 8; ++v) {
        int row = v + ((lane >> 4) << 3);
        sO[(mt * 16 + row) * 128 + (nh * 4 + j) * 16 + (lane & 15)] =
            fmaxf(acc[j][v], 0.f);
      }
    __syncthreads();

    for (int i = 0; i < 8; ++i) {  // coalesced f32 store
      int g = t + i * 256;
      int node = base + (g >> 5);
      if (node < NN)
        ((float4*)x2)[(size_t)node * 32 + (g & 31)] = ((const float4*)sO)[g];
    }
    __syncthreads();
  }
}

// =====================================================================
// Kernel 4: 2x2 max-pool (closed-form NEW_NODE_IDS) + new_pos
// =====================================================================
__global__ __launch_bounds__(256) void k_pool(const float* __restrict__ x2,
                                              float* __restrict__ outx,
                                              int* __restrict__ outpos) {
  int t = blockIdx.x * 256 + threadIdx.x;
  if (t >= NP * 128) return;
  int p = t >> 7, c = t & 127;
  int pr = p / PW, pc = p % PW;
  int r0 = pr * 2, c0 = pc * 2;
  const float* base = x2 + ((size_t)(r0 * GW + c0)) * 128 + c;
  float v = fmaxf(base[0], base[128]);
  if (r0 + 1 < GH) {
    const float* b2 = base + (size_t)GW * 128;
    v = fmaxf(v, fmaxf(b2[0], b2[128]));
  }
  outx[(size_t)p * 128 + c] = v;
  if (c < 2) outpos[(size_t)p * 2 + c] = (c == 0) ? pr : pc;
}

// =====================================================================
// Kernel 5: pooled-edge dedup outputs (closed-form groups on the grid)
// =====================================================================
__global__ __launch_bounds__(256) void k_edgeout(const float* __restrict__ qe,
                                                 float* __restrict__ ef_out,
                                                 int* __restrict__ ei_out,
                                                 int Kkeep, int dir) {
  int t = blockIdx.x * 256 + threadIdx.x;
  if (t >= Kkeep * 32) return;
  int k = t >> 5, u = t & 31;
  int e0, e1, s, d;
  if (dir == 0) {            // north: kept pair (b+1 -> b), group rr=2(b+1)
    int b = k / PW, c2 = k % PW;
    e0 = (2 * (b + 1) - 1) * GW + 2 * c2;
    e1 = e0 + 1;
    s = (b + 1) * PW + c2;
    d = b * PW + c2;
  } else if (dir == 2) {     // south: kept pair (b -> b+1), group rr=2b+1
    int b = k / PW, c2 = k % PW;
    e0 = (2 * b + 1) * GW + 2 * c2;
    e1 = e0 + 1;
    s = b * PW + c2;
    d = (b + 1) * PW + c2;
  } else if (dir == 1) {     // east: kept pair (j -> j+1), group cc=2j+1
    int pr = k / 217, j = k % 217;
    e0 = (2 * pr) * (GW - 1) + 2 * j + 1;
    e1 = (2 * pr + 1 < GH) ? e0 + (GW - 1) : -1;
    s = pr * PW + j;
    d = s + 1;
  } else {                   // west: kept pair (j -> j-1), group cc=2j
    int pr = k / 217, j = k % 217 + 1;
    e0 = (2 * pr) * (GW - 1) + (2 * j - 1);
    e1 = (2 * pr + 1 < GH) ? e0 + (GW - 1) : -1;
    s = pr * PW + j;
    d = s - 1;
  }
  float v = qe[(size_t)e0 * 32 + u];
  if (e1 >= 0) v = fmaxf(v, qe[(size_t)e1 * 32 + u]);
  ef_out[(size_t)k * 32 + u] = v;
  if (u == 0) {
    ei_out[(size_t)k * 2 + 0] = s;
    ei_out[(size_t)k * 2 + 1] = d;
  }
}

// =====================================================================
extern "C" void kernel_launch(void* const* d_in, const int* in_sizes, int n_in,
                              void* d_out, int out_size, void* d_ws,
                              size_t ws_size, hipStream_t stream) {
  const float* x    = (const float*)d_in[0];
  const int*   ein  = (const int*)d_in[2];
  const int*   eie  = (const int*)d_in[3];
  const int*   eis  = (const int*)d_in[4];
  const int*   eiw  = (const int*)d_in[5];
  const float* efn  = (const float*)d_in[6];
  const float* efe  = (const float*)d_in[7];
  const float* efs  = (const float*)d_in[8];
  const float* efw  = (const float*)d_in[9];
  const float* Wemb = (const float*)d_in[10];
  const float* Wedg = (const float*)d_in[11];
  const float* Wnod = (const float*)d_in[12];

  char* ws = (char*)d_ws;
  u16*   xq  = (u16*)(ws + WS_XQ);
  u16*   agg = (u16*)(ws + WS_AGG);
  float* x2  = (float*)(ws + WS_X2);
  float* qn  = (float*)(ws + WS_QN);
  float* qe  = (float*)(ws + WS_QE);
  float* qs  = (float*)(ws + WS_QS);
  float* qw  = (float*)(ws + WS_QW);

  (void)hipMemsetAsync(agg, 0, (size_t)NN * 128 * 2, stream);

  k_embed<<<512, 256, 0, stream>>>(x, Wemb, xq);

  k_edge<<<512, 256, 0, stream>>>(efn, ein, xq, Wedg + 0 * 96 * 32, agg, qn, EN, 0, (EN + 63) / 64);
  k_edge<<<512, 256, 0, stream>>>(efe, eie, xq, Wedg + 1 * 96 * 32, agg, qe, EE, 1, (EE + 63) / 64);
  k_edge<<<512, 256, 0, stream>>>(efs, eis, xq, Wedg + 2 * 96 * 32, agg, qs, ES, 2, (ES + 63) / 64);
  k_edge<<<512, 256, 0, stream>>>(efw, eiw, xq, Wedg + 3 * 96 * 32, agg, qw, EW, 3, (EW + 63) / 64);

  k_node<<<512, 256, 0, stream>>>(x, agg, Wnod, x2);

  float* outf = (float*)d_out;
  k_pool<<<(NP * 128 + 255) / 256, 256, 0, stream>>>(x2, outf + O_XP,
                                                     (int*)(outf + O_POS));

  k_edgeout<<<(KN * 32 + 255) / 256, 256, 0, stream>>>(qn, outf + O_EFN, (int*)(outf + O_EIN), KN, 0);
  k_edgeout<<<(KE * 32 + 255) / 256, 256, 0, stream>>>(qe, outf + O_EFE, (int*)(outf + O_EIE), KE, 1);
  k_edgeout<<<(KS * 32 + 255) / 256, 256, 0, stream>>>(qs, outf + O_EFS, (int*)(outf + O_EIS), KS, 2);
  k_edgeout<<<(KW * 32 + 255) / 256, 256, 0, stream>>>(qw, outf + O_EFW, (int*)(outf + O_EIW), KW, 3);
}
